// MultimodalGNN_54503134986515
// MI455X (gfx1250) — matrix operand order repaired and checked by reading, not compiled, not run on hardware
//
#include <hip/hip_runtime.h>
#include <hip/hip_bf16.h>

typedef __attribute__((ext_vector_type(16))) _Float16 v16h;
typedef __attribute__((ext_vector_type(8)))  float    v8f;
typedef __attribute__((ext_vector_type(4)))  float    v4f;

#define EPSV 1e-5f

// ---------------------------------------------------------------------------
// Stage 1: per-node conv stack (1->16->32->64) + BN + ReLU + mean pool.
// One 256-thread block per node, activations staged in LDS (~26 KB).
// ---------------------------------------------------------------------------
__global__ __launch_bounds__(256) void conv_stack_kernel(
    const float* __restrict__ x,
    const float* __restrict__ w1, const float* __restrict__ cb1,
    const float* __restrict__ w2, const float* __restrict__ cb2,
    const float* __restrict__ w3, const float* __restrict__ cb3,
    const float* __restrict__ g1, const float* __restrict__ bb1,
    const float* __restrict__ m1, const float* __restrict__ v1,
    const float* __restrict__ g2, const float* __restrict__ bb2,
    const float* __restrict__ m2, const float* __restrict__ v2,
    const float* __restrict__ g3, const float* __restrict__ bb3,
    const float* __restrict__ m3, const float* __restrict__ v3,
    float* __restrict__ h)
{
    const int node = blockIdx.x;
    const int tid  = threadIdx.x;

    __shared__ float xs[256];
    __shared__ float a1[16 * 128];
    __shared__ float a2[32 * 64];
    __shared__ float a3[64 * 32];

    xs[tid] = x[(long)node * 256 + tid];
    __syncthreads();

    // conv1: out 16 x 128, k=7 s=2 p=3
    for (int idx = tid; idx < 16 * 128; idx += 256) {
        const int o = idx >> 7, p = idx & 127;
        float acc = cb1[o];
        const int base = 2 * p - 3;
#pragma unroll
        for (int t = 0; t < 7; ++t) {
            const int q = base + t;
            if (q >= 0 && q < 256) acc += xs[q] * w1[o * 7 + t];
        }
        const float sc = g1[o] * rsqrtf(v1[o] + EPSV);
        const float val = (acc - m1[o]) * sc + bb1[o];
        a1[idx] = val > 0.f ? val : 0.f;
    }
    __syncthreads();

    // conv2: out 32 x 64, in 16ch, k=5 s=2 p=2
    for (int idx = tid; idx < 32 * 64; idx += 256) {
        const int o = idx >> 6, p = idx & 63;
        float acc = cb2[o];
        const int base = 2 * p - 2;
        for (int ci = 0; ci < 16; ++ci) {
            const float* wr = w2 + (o * 16 + ci) * 5;
            const float* ar = a1 + ci * 128;
#pragma unroll
            for (int t = 0; t < 5; ++t) {
                const int q = base + t;
                if (q >= 0 && q < 128) acc += ar[q] * wr[t];
            }
        }
        const float sc = g2[o] * rsqrtf(v2[o] + EPSV);
        const float val = (acc - m2[o]) * sc + bb2[o];
        a2[idx] = val > 0.f ? val : 0.f;
    }
    __syncthreads();

    // conv3: out 64 x 32, in 32ch, k=3 s=2 p=1
    for (int idx = tid; idx < 64 * 32; idx += 256) {
        const int o = idx >> 5, p = idx & 31;
        float acc = cb3[o];
        const int base = 2 * p - 1;
        for (int ci = 0; ci < 32; ++ci) {
            const float* wr = w3 + (o * 32 + ci) * 3;
            const float* ar = a2 + ci * 64;
#pragma unroll
            for (int t = 0; t < 3; ++t) {
                const int q = base + t;
                if (q >= 0 && q < 64) acc += ar[q] * wr[t];
            }
        }
        const float sc = g3[o] * rsqrtf(v3[o] + EPSV);
        const float val = (acc - m3[o]) * sc + bb3[o];
        a3[idx] = val > 0.f ? val : 0.f;
    }
    __syncthreads();

    if (tid < 64) {
        float s = 0.f;
#pragma unroll
        for (int p = 0; p < 32; ++p) s += a3[tid * 32 + p];
        h[(long)node * 64 + tid] = s * (1.f / 32.f);
    }
}

// ---------------------------------------------------------------------------
// Small utility kernels
// ---------------------------------------------------------------------------
__global__ void fill_kernel(float* __restrict__ p, float v, long n) {
    const long i = (long)blockIdx.x * 256 + threadIdx.x;
    if (i < n) p[i] = v;
}

__global__ void deg_accum_kernel(const int* __restrict__ col,
                                 float* __restrict__ deg, int e) {
    const int i = blockIdx.x * 256 + threadIdx.x;
    if (i < e) atomicAdd(deg + col[i], 1.0f);
}

__global__ void dinv_kernel(float* __restrict__ d, int n) {
    const int i = blockIdx.x * 256 + threadIdx.x;
    if (i < n) d[i] = rsqrtf(d[i]);
}

// ---------------------------------------------------------------------------
// Pack an f32 weight matrix W[64][C] (C = 16*CT columns) into f16 WMMA
// B-operand lane order: dst[ct*1024 + kb*512 + lane*16 + j] with
// k = 32*kb + 8*(lane>>4) + j (j<8) / 16 + 8*(lane>>4) + j-8 (j>=8), n = 16*ct + (lane&15). Each lane's operand is
// then 32 contiguous bytes -> single vector load in the GEMM.
// ---------------------------------------------------------------------------
__global__ void pack_b_kernel(const float* __restrict__ w,
                              _Float16* __restrict__ bp, int C, int total) {
    const int idx = blockIdx.x * 256 + threadIdx.x;
    if (idx >= total) return;
    const int j    = idx & 15;
    const int lane = (idx >> 4) & 31;
    const int kb   = (idx >> 9) & 1;
    const int ct   = idx >> 10;
    const int k = 32 * kb + (j < 8 ? 8 * (lane >> 4) + j : 16 + 8 * (lane >> 4) + (j - 8));
    const int n = 16 * ct + (lane & 15);
    bp[idx] = (_Float16)w[k * C + n];
}

// ---------------------------------------------------------------------------
// WMMA GEMM: out[M, 16*CT] = h[M,64] @ W[64, 16*CT]  (f16 ops, f32 accum)
// 256-thread block = 8 waves; each wave owns one 16-row stripe and all
// columns: CT column tiles x 2 K-steps of v_wmma_f32_16x16x32_f16.
// B comes pre-packed in lane order (32B contiguous per lane per operand).
// ---------------------------------------------------------------------------
template <int CT>
__global__ __launch_bounds__(256) void gemm_wmma_kernel(
    const float* __restrict__ h, const _Float16* __restrict__ Bp,
    float* __restrict__ out, int n)
{
    const int wave = threadIdx.x >> 5;
    const int lane = threadIdx.x & 31;
    const int m    = lane & 15;   // row-in-tile (A) / col-in-tile (C)
    const int kh   = lane >> 4;   // K half selector

    const long row0 = ((long)blockIdx.x * 8 + wave) * 16;
    const long r    = row0 + m;
    const bool valid = (r < n);
    const float* hr = h + (valid ? r : 0) * 64;

    // A operands: 16x32 f16. Per lane the K map is two contiguous 8-float
    // runs: [8*kh, 8*kh+8) and [16+8*kh, 16+8*kh+8); +32 for the second step.
    v16h a0, a1;
    if (valid) {
        const v4f x0 = *(const v4f*)(hr + 8 * kh);
        const v4f x1 = *(const v4f*)(hr + 8 * kh + 4);
        const v4f x2 = *(const v4f*)(hr + 16 + 8 * kh);
        const v4f x3 = *(const v4f*)(hr + 20 + 8 * kh);
        const v4f y0 = *(const v4f*)(hr + 32 + 8 * kh);
        const v4f y1 = *(const v4f*)(hr + 36 + 8 * kh);
        const v4f y2 = *(const v4f*)(hr + 48 + 8 * kh);
        const v4f y3 = *(const v4f*)(hr + 52 + 8 * kh);
#pragma unroll
        for (int j = 0; j < 4; ++j) {
            a0[j]      = (_Float16)x0[j];
            a0[j + 4]  = (_Float16)x1[j];
            a0[j + 8]  = (_Float16)x2[j];
            a0[j + 12] = (_Float16)x3[j];
            a1[j]      = (_Float16)y0[j];
            a1[j + 4]  = (_Float16)y1[j];
            a1[j + 8]  = (_Float16)y2[j];
            a1[j + 12] = (_Float16)y3[j];
        }
    } else {
#pragma unroll
        for (int j = 0; j < 16; ++j) { a0[j] = (_Float16)0.f; a1[j] = (_Float16)0.f; }
    }

    // B operands: one 32-byte contiguous vector load per (ct, kb).
    v16h bm[CT][2];
#pragma unroll
    for (int ct = 0; ct < CT; ++ct) {
#pragma unroll
        for (int kb = 0; kb < 2; ++kb) {
            bm[ct][kb] = *(const v16h*)(Bp + (ct * 2 + kb) * 512 + lane * 16);
        }
    }

    const v8f vzero = {0.f, 0.f, 0.f, 0.f, 0.f, 0.f, 0.f, 0.f};
    v8f acc[CT];
#pragma unroll
    for (int ct = 0; ct < CT; ++ct) acc[ct] = vzero;

#pragma unroll
    for (int ct = 0; ct < CT; ++ct) {
        acc[ct] = __builtin_amdgcn_wmma_f32_16x16x32_f16(
            false, a0, false, bm[ct][0], (short)0, acc[ct], false, false);
        acc[ct] = __builtin_amdgcn_wmma_f32_16x16x32_f16(
            false, a1, false, bm[ct][1], (short)0, acc[ct], false, false);
    }

    // C/D layout: VGPR v, lanes 0-15 -> M=v, lanes 16-31 -> M=v+8.
    // Hot path: full tile in range (N is a multiple of 16) -> no guards.
    if (row0 + 16 <= n) {
#pragma unroll
        for (int ct = 0; ct < CT; ++ct) {
#pragma unroll
            for (int v = 0; v < 8; ++v) {
                const long rr = row0 + v + 8 * kh;
                out[rr * (CT * 16) + 16 * ct + m] = acc[ct][v];
            }
        }
    } else {
#pragma unroll
        for (int ct = 0; ct < CT; ++ct) {
#pragma unroll
            for (int v = 0; v < 8; ++v) {
                const long rr = row0 + v + 8 * kh;
                if (rr < n) out[rr * (CT * 16) + 16 * ct + m] = acc[ct][v];
            }
        }
    }
}

// ---------------------------------------------------------------------------
// GCN propagation (float4-granular: one thread = 4 channels)
// ---------------------------------------------------------------------------
__global__ void selfloop_kernel(const v4f* __restrict__ xw4,
                                const float* __restrict__ dinv,
                                v4f* __restrict__ out4, long n16) {
    const long idx = (long)blockIdx.x * 256 + threadIdx.x;
    if (idx >= n16) return;
    const float dv = dinv[idx >> 4];
    out4[idx] = xw4[idx] * (dv * dv);
}

__global__ void edge_prop_kernel(const int* __restrict__ row,
                                 const int* __restrict__ col,
                                 const v4f* __restrict__ xw4,
                                 const float* __restrict__ dinv,
                                 float* __restrict__ out, long total) {
    const long t = (long)blockIdx.x * 256 + threadIdx.x;
    if (t >= total) return;                 // total = E * 16
    const long e  = t >> 4;
    const int  c4 = (int)(t & 15);
    const int  r  = row[e];
    const int  cl = col[e];
    __builtin_prefetch((const float*)xw4 + (long)r * 64 + c4 * 4, 0, 0);
    const float nrm = dinv[r] * dinv[cl];
    const v4f v = xw4[(long)r * 16 + c4] * nrm;
    float* dst = out + (long)cl * 64 + c4 * 4;
#pragma unroll
    for (int k = 0; k < 4; ++k) atomicAdd(dst + k, v[k]);
}

// total4 = elements/4 ; cm4 = (channels/4) - 1
__global__ void bias_relu_kernel(v4f* __restrict__ h4,
                                 const v4f* __restrict__ b4,
                                 long total4, int cm4) {
    const long idx = (long)blockIdx.x * 256 + threadIdx.x;
    if (idx >= total4) return;
    v4f v = h4[idx] + b4[idx & cm4];
#pragma unroll
    for (int k = 0; k < 4; ++k) v[k] = v[k] > 0.f ? v[k] : 0.f;
    h4[idx] = v;
}

// ---------------------------------------------------------------------------
// Graph pooling (float4-granular)
// ---------------------------------------------------------------------------
__global__ void pool_accum_kernel(const v4f* __restrict__ h4,
                                  const int* __restrict__ batch,
                                  float* __restrict__ sums,
                                  float* __restrict__ cnts, long n16) {
    const long idx = (long)blockIdx.x * 256 + threadIdx.x;
    if (idx >= n16) return;
    const long i  = idx >> 4;
    const int  c4 = (int)(idx & 15);
    const int  g  = batch[i];
    const v4f v = h4[idx];
    float* dst = sums + (long)g * 64 + c4 * 4;
#pragma unroll
    for (int k = 0; k < 4; ++k) atomicAdd(dst + k, v[k]);
    if (c4 == 0) atomicAdd(cnts + g, 1.0f);
}

__global__ void ge_kernel(const float* __restrict__ sums,
                          const float* __restrict__ cnts,
                          float* __restrict__ ge, int g64) {
    const int idx = blockIdx.x * 256 + threadIdx.x;
    if (idx >= g64) return;
    const float cc = cnts[idx >> 6];
    ge[idx] = sums[idx] / fmaxf(cc, 1.0f);
}

// ---------------------------------------------------------------------------
// Final head projections from WMMA-produced hidden activations:
// logits[g,0:2] = hiddenC[g,:]@W2c + b2c ; mag[g] = hiddenM[g,:]@W2m + b2m
// ---------------------------------------------------------------------------
__global__ __launch_bounds__(64) void heads_final_kernel(
    const float* __restrict__ hc, const float* __restrict__ w2c,
    const float* __restrict__ b2c,
    const float* __restrict__ hm, const float* __restrict__ w2m,
    const float* __restrict__ b2m,
    float* __restrict__ out, int G)
{
    const int g = blockIdx.x;
    const int t = threadIdx.x;
    if (t < 2) {
        float s = b2c[t];
        for (int k = 0; k < 32; ++k) s += hc[(long)g * 32 + k] * w2c[k * 2 + t];
        out[g * 2 + t] = s;                 // logits [G,2]
    } else if (t == 2) {
        float s = b2m[0];
        for (int k = 0; k < 32; ++k) s += hm[(long)g * 32 + k] * w2m[k];
        out[G * 2 + g] = s;                 // mag [G,1] after logits
    }
}

// ---------------------------------------------------------------------------
// Launch
// ---------------------------------------------------------------------------
static inline unsigned nblk(long n, int b) { return (unsigned)((n + b - 1) / b); }

extern "C" void kernel_launch(void* const* d_in, const int* in_sizes, int n_in,
                              void* d_out, int out_size, void* d_ws, size_t ws_size,
                              hipStream_t stream) {
    (void)n_in; (void)out_size; (void)ws_size;

    const long N = in_sizes[2];          // batch has N entries
    const long E = in_sizes[1] / 2;      // edge_index is [2,E]
    const int  G = 256;

    const float* x       = (const float*)d_in[0];
    const int*   ei      = (const int*)d_in[1];
    const int*   batch   = (const int*)d_in[2];
    const float* conv1_w = (const float*)d_in[3];
    const float* conv1_b = (const float*)d_in[4];
    const float* conv2_w = (const float*)d_in[5];
    const float* conv2_b = (const float*)d_in[6];
    const float* conv3_w = (const float*)d_in[7];
    const float* conv3_b = (const float*)d_in[8];
    const float* bn1_g = (const float*)d_in[9],  *bn1_b = (const float*)d_in[10];
    const float* bn1_m = (const float*)d_in[11], *bn1_v = (const float*)d_in[12];
    const float* bn2_g = (const float*)d_in[13], *bn2_b = (const float*)d_in[14];
    const float* bn2_m = (const float*)d_in[15], *bn2_v = (const float*)d_in[16];
    const float* bn3_g = (const float*)d_in[17], *bn3_b = (const float*)d_in[18];
    const float* bn3_m = (const float*)d_in[19], *bn3_v = (const float*)d_in[20];
    const float* gcn1_w = (const float*)d_in[21], *gcn1_b = (const float*)d_in[22];
    const float* gcn2_w = (const float*)d_in[23], *gcn2_b = (const float*)d_in[24];
    const float* clf_w1 = (const float*)d_in[25], *clf_b1 = (const float*)d_in[26];
    const float* clf_w2 = (const float*)d_in[27], *clf_b2 = (const float*)d_in[28];
    const float* mag_w1 = (const float*)d_in[29], *mag_b1 = (const float*)d_in[30];
    const float* mag_w2 = (const float*)d_in[31], *mag_b2 = (const float*)d_in[32];

    const int* rowi = ei;
    const int* coli = ei + E;

    // workspace carve-out
    char*  ws  = (char*)d_ws;
    size_t off = 0;
    auto alloc = [&](size_t bytes) {
        off = (off + 255) & ~(size_t)255;
        size_t cur = off;
        off += bytes;
        return cur;
    };
    float*    bufA  = (float*)(ws + alloc((size_t)N * 64 * 4));
    float*    bufB  = (float*)(ws + alloc((size_t)N * 64 * 4));
    float*    bufC  = (float*)(ws + alloc((size_t)N * 64 * 4));
    float*    dinv  = (float*)(ws + alloc((size_t)N * 4));
    _Float16* BpG   = (_Float16*)(ws + alloc(4 * 1024 * 2));   // GCN  (CT=4)
    _Float16* BpC   = (_Float16*)(ws + alloc(2 * 1024 * 2));   // clf  (CT=2)
    _Float16* BpM   = (_Float16*)(ws + alloc(2 * 1024 * 2));   // mag  (CT=2)
    float*    sums  = (float*)(ws + alloc((size_t)G * 64 * 4));
    float*    cnts  = (float*)(ws + alloc((size_t)G * 4));
    float*    ge    = (float*)(ws + alloc((size_t)G * 64 * 4));
    float*    hidC  = (float*)(ws + alloc((size_t)G * 32 * 4));
    float*    hidM  = (float*)(ws + alloc((size_t)G * 32 * 4));

    const long n16    = N * 16;       // float4 elements per N x 64 buffer
    const long etotal = E * 16;       // float4 tasks for edge scatter

    // 1) conv stack -> bufA = h[N,64]
    conv_stack_kernel<<<(unsigned)N, 256, 0, stream>>>(
        x, conv1_w, conv1_b, conv2_w, conv2_b, conv3_w, conv3_b,
        bn1_g, bn1_b, bn1_m, bn1_v, bn2_g, bn2_b, bn2_m, bn2_v,
        bn3_g, bn3_b, bn3_m, bn3_v, bufA);

    // 2) degree (self loop = 1) -> dinv
    fill_kernel<<<nblk(N, 256), 256, 0, stream>>>(dinv, 1.0f, N);
    deg_accum_kernel<<<nblk(E, 256), 256, 0, stream>>>(coli, dinv, (int)E);
    dinv_kernel<<<nblk(N, 256), 256, 0, stream>>>(dinv, (int)N);

    // 3) GCN layer 1: bufA -> bufC
    pack_b_kernel<<<nblk(4096, 256), 256, 0, stream>>>(gcn1_w, BpG, 64, 4096);
    gemm_wmma_kernel<4><<<nblk(N, 128), 256, 0, stream>>>(bufA, BpG, bufB, (int)N);
    selfloop_kernel<<<nblk(n16, 256), 256, 0, stream>>>((const v4f*)bufB, dinv, (v4f*)bufC, n16);
    edge_prop_kernel<<<nblk(etotal, 256), 256, 0, stream>>>(rowi, coli, (const v4f*)bufB, dinv, bufC, etotal);
    bias_relu_kernel<<<nblk(n16, 256), 256, 0, stream>>>((v4f*)bufC, (const v4f*)gcn1_b, n16, 15);

    // 4) GCN layer 2: bufC -> bufA
    pack_b_kernel<<<nblk(4096, 256), 256, 0, stream>>>(gcn2_w, BpG, 64, 4096);
    gemm_wmma_kernel<4><<<nblk(N, 128), 256, 0, stream>>>(bufC, BpG, bufB, (int)N);
    selfloop_kernel<<<nblk(n16, 256), 256, 0, stream>>>((const v4f*)bufB, dinv, (v4f*)bufA, n16);
    edge_prop_kernel<<<nblk(etotal, 256), 256, 0, stream>>>(rowi, coli, (const v4f*)bufB, dinv, bufA, etotal);
    bias_relu_kernel<<<nblk(n16, 256), 256, 0, stream>>>((v4f*)bufA, (const v4f*)gcn2_b, n16, 15);

    // 5) pooling -> ge[G,64]
    fill_kernel<<<nblk((long)G * 64, 256), 256, 0, stream>>>(sums, 0.0f, (long)G * 64);
    fill_kernel<<<nblk(G, 256), 256, 0, stream>>>(cnts, 0.0f, G);
    pool_accum_kernel<<<nblk(n16, 256), 256, 0, stream>>>((const v4f*)bufA, batch, sums, cnts, n16);
    ge_kernel<<<nblk((long)G * 64, 256), 256, 0, stream>>>(sums, cnts, ge, G * 64);

    // 6) head hidden layers via WMMA: hid = relu(ge @ W1 + b1)
    pack_b_kernel<<<nblk(2048, 256), 256, 0, stream>>>(clf_w1, BpC, 32, 2048);
    pack_b_kernel<<<nblk(2048, 256), 256, 0, stream>>>(mag_w1, BpM, 32, 2048);
    gemm_wmma_kernel<2><<<nblk(G, 128), 256, 0, stream>>>(ge, BpC, hidC, G);
    gemm_wmma_kernel<2><<<nblk(G, 128), 256, 0, stream>>>(ge, BpM, hidM, G);
    bias_relu_kernel<<<nblk((long)G * 8, 256), 256, 0, stream>>>((v4f*)hidC, (const v4f*)clf_b1, (long)G * 8, 7);
    bias_relu_kernel<<<nblk((long)G * 8, 256), 256, 0, stream>>>((v4f*)hidM, (const v4f*)mag_b1, (long)G * 8, 7);

    // 7) final projections -> d_out (logits [G,2] then mag [G,1], flat)
    heads_final_kernel<<<G, 64, 0, stream>>>(hidC, clf_w2, clf_b2,
                                             hidM, mag_w2, mag_b2,
                                             (float*)d_out, G);
}
